// WeightOnlyBlockwiseQuantizedLinear_38225208934698
// MI455X (gfx1250) — compile-verified, matrix operand order checked by reading, not run on hardware
//
#include <hip/hip_runtime.h>
#include <hip/hip_bf16.h>

// Problem constants (fixed by the reference).
constexpr int M    = 2 * 1024;  // B*D
constexpr int K    = 4096;      // IN_F
constexpr int N    = 4096;      // OUT_F
constexpr int QBLK = 128;       // quantization block in K
constexpr int NBLK = K / QBLK;  // 32

// GEMM tiling.
constexpr int BM = 64;
constexpr int BN = 128;
constexpr int BK = QBLK;        // K tile == one quant block
constexpr int XSTR = BK + 8;    // LDS strides (halves); rows stay 16B aligned
constexpr int WSTR = BK + 8;

typedef __attribute__((ext_vector_type(16))) _Float16 v16h;
typedef __attribute__((ext_vector_type(8)))  _Float16 v8h;
typedef __attribute__((ext_vector_type(4)))  _Float16 v4h;
typedef __attribute__((ext_vector_type(8)))  float    v8f;
typedef __attribute__((ext_vector_type(4)))  float    v4f;
typedef __attribute__((ext_vector_type(4)))  int      v4i;

#if defined(__AMDGCN__) && __has_builtin(__builtin_amdgcn_global_load_async_to_lds_b128)
#define HAVE_ASYNC_LDS 1
#else
#define HAVE_ASYNC_LDS 0
#endif

#if HAVE_ASYNC_LDS
typedef __attribute__((address_space(1))) v4i* g_v4i_p;  // global int4*
typedef __attribute__((address_space(3))) v4i* l_v4i_p;  // LDS int4*
#endif

static __device__ inline void wait_async_zero() {
#if defined(__AMDGCN__)
#if __has_builtin(__builtin_amdgcn_s_wait_asynccnt)
    __builtin_amdgcn_s_wait_asynccnt(0);
#else
    asm volatile("s_wait_asynccnt 0x0" ::: "memory");
#endif
#endif
}

static __device__ inline v16h concat8(v8h lo, v8h hi) {
    v16h r;
#pragma unroll
    for (int i = 0; i < 8; ++i) { r[i] = lo[i]; r[i + 8] = hi[i]; }
    return r;
}

// ---------------------------------------------------------------------------
// Pass 1a: X f32 [M,K] -> f16 [M,K]
// ---------------------------------------------------------------------------
__global__ __launch_bounds__(256)
void cvt_x_kernel(const float* __restrict__ x, _Float16* __restrict__ xh) {
    const int t = threadIdx.x;
    const size_t base = (size_t)blockIdx.x * 1024;  // float4 chunks
#pragma unroll
    for (int i = 0; i < 4; ++i) {
        const size_t c4 = base + i * 256 + t;
        const v4f v = *(const v4f*)(x + c4 * 4);
        v4h h;
#pragma unroll
        for (int j = 0; j < 4; ++j) h[j] = (_Float16)v[j];
        *(v4h*)(xh + c4 * 4) = h;
    }
}

// ---------------------------------------------------------------------------
// Pass 1b: W int8 [K,N] -> f16 transposed [N,K], via LDS tile transpose.
// Exact: |int8| <= 128 fits in f16 significand.
// ---------------------------------------------------------------------------
__global__ __launch_bounds__(256)
void cvt_w_kernel(const signed char* __restrict__ wq, _Float16* __restrict__ whT) {
    __shared__ _Float16 T[128 * 136];  // [n][k]
    const int tid = threadIdx.x;
    const int k0  = blockIdx.x * 128;
    const int n0  = blockIdx.y * 128;

    // Read coalesced along n, scatter into LDS transposed.
#pragma unroll
    for (int i = 0; i < 16; ++i) {
        const int idx = tid + i * 256;
        const int k   = idx >> 5;
        const int n4  = (idx & 31) * 4;
        const unsigned w32 =
            *(const unsigned*)(wq + (size_t)(k0 + k) * N + n0 + n4);
#pragma unroll
        for (int j = 0; j < 4; ++j) {
            const signed char b = (signed char)((w32 >> (8 * j)) & 0xff);
            T[(n4 + j) * 136 + k] = (_Float16)b;
        }
    }
    __syncthreads();

    // Write coalesced along k.
#pragma unroll
    for (int i = 0; i < 8; ++i) {
        const int idx = tid + i * 256;   // 2048 chunks of 8 halves
        const int n   = idx >> 4;
        const int c8  = (idx & 15) * 8;
        *(v8h*)(whT + (size_t)(n0 + n) * K + k0 + c8) = *(const v8h*)&T[n * 136 + c8];
    }
}

// ---------------------------------------------------------------------------
// Pass 2: GEMM on pre-converted f16 operands. Staging is a pure copy:
// async global->LDS when available, otherwise 128-bit vector copies.
// ---------------------------------------------------------------------------
__global__ __launch_bounds__(256)
void gemm_f16_kernel(const _Float16* __restrict__ xh,   // [M,K]
                     const _Float16* __restrict__ whT,  // [N,K]
                     const float* __restrict__ scal,    // [NBLK,N]
                     float* __restrict__ out)           // [M,N]
{
    __shared__ _Float16 Xs[BM * XSTR];
    __shared__ _Float16 Ws[BN * WSTR];

    const int tid    = threadIdx.x;
    const int lane   = tid & 31;          // wave32
    const int wid    = tid >> 5;
    const int wave_m = wid >> 2;          // 0..1
    const int wave_n = wid & 3;           // 0..3
    const int l15    = lane & 15;
    const int lhalf  = (lane >> 4) & 1;
    const int koff   = lhalf * 8;

    const int bm = blockIdx.y * BM;
    const int bn = blockIdx.x * BN;

    v8f facc[2][2];
#pragma unroll
    for (int mi = 0; mi < 2; ++mi)
#pragma unroll
        for (int tn = 0; tn < 2; ++tn)
            facc[mi][tn] = (v8f){0.f, 0.f, 0.f, 0.f, 0.f, 0.f, 0.f, 0.f};

    for (int s = 0; s < NBLK; ++s) {
        const int kb = s * BK;

        float sc[2];
#pragma unroll
        for (int tn = 0; tn < 2; ++tn)
            sc[tn] = scal[s * N + bn + wave_n * 32 + tn * 16 + l15];

        __syncthreads();  // previous compute done before LDS overwrite

        // ---- Stage X tile 64x128 halves: 1024 x 16B chunks, 4/thread ----
#pragma unroll
        for (int i = 0; i < 4; ++i) {
            const int idx = tid + i * 256;
            const int row = idx >> 4;
            const int c8  = (idx & 15) * 8;
            const _Float16* src = xh + (size_t)(bm + row) * K + kb + c8;
            _Float16*       dst = &Xs[row * XSTR + c8];
#if HAVE_ASYNC_LDS
            __builtin_amdgcn_global_load_async_to_lds_b128((g_v4i_p)src, (l_v4i_p)dst, 0, 0);
#else
            *(v8h*)dst = *(const v8h*)src;
#endif
        }

        // ---- Stage W tile 128x128 halves: 2048 x 16B chunks, 8/thread ----
#pragma unroll
        for (int i = 0; i < 8; ++i) {
            const int idx = tid + i * 256;
            const int n   = idx >> 4;
            const int c8  = (idx & 15) * 8;
            const _Float16* src = whT + (size_t)(bn + n) * K + kb + c8;
            _Float16*       dst = &Ws[n * WSTR + c8];
#if HAVE_ASYNC_LDS
            __builtin_amdgcn_global_load_async_to_lds_b128((g_v4i_p)src, (l_v4i_p)dst, 0, 0);
#else
            *(v8h*)dst = *(const v8h*)src;
#endif
        }

        // Prefetch next k-block (global_prefetch_b8).
        if (s + 1 < NBLK) {
            const int kn = kb + BK;
            __builtin_prefetch(xh + (size_t)(bm + (tid >> 2)) * K + kn + (tid & 3) * 32, 0, 1);
            __builtin_prefetch(whT + (size_t)(bn + (tid >> 1)) * K + kn + (tid & 1) * 64, 0, 1);
        }

#if HAVE_ASYNC_LDS
        wait_async_zero();
#endif
        __syncthreads();

        v8f bacc[2][2];
#pragma unroll
        for (int mi = 0; mi < 2; ++mi)
#pragma unroll
            for (int tn = 0; tn < 2; ++tn)
                bacc[mi][tn] = (v8f){0.f, 0.f, 0.f, 0.f, 0.f, 0.f, 0.f, 0.f};

#pragma unroll
        for (int ks = 0; ks < 4; ++ks) {
            const int kk = ks * 32;

            v16h A[2];
#pragma unroll
            for (int mi = 0; mi < 2; ++mi) {
                const int row  = wave_m * 32 + mi * 16 + l15;
                const int base = row * XSTR + kk + koff;
                A[mi] = concat8(*(const v8h*)&Xs[base], *(const v8h*)&Xs[base + 16]);
            }
            v16h Bf[2];
#pragma unroll
            for (int tn = 0; tn < 2; ++tn) {
                const int col  = wave_n * 32 + tn * 16 + l15;
                const int base = col * WSTR + kk + koff;
                Bf[tn] = concat8(*(const v8h*)&Ws[base], *(const v8h*)&Ws[base + 16]);
            }
#pragma unroll
            for (int mi = 0; mi < 2; ++mi)
#pragma unroll
                for (int tn = 0; tn < 2; ++tn)
                    bacc[mi][tn] = __builtin_amdgcn_wmma_f32_16x16x32_f16(
                        false, A[mi], false, Bf[tn],
                        (short)0, bacc[mi][tn], false, false);
        }

#pragma unroll
        for (int mi = 0; mi < 2; ++mi)
#pragma unroll
            for (int tn = 0; tn < 2; ++tn) {
                const float s_ = sc[tn];
#pragma unroll
                for (int e = 0; e < 8; ++e)
                    facc[mi][tn][e] = fmaf(s_, bacc[mi][tn][e], facc[mi][tn][e]);
            }
    }

    const int mrow0 = bm + wave_m * 32 + lhalf * 8;
    const int ncol0 = bn + wave_n * 32 + l15;
#pragma unroll
    for (int mi = 0; mi < 2; ++mi)
#pragma unroll
        for (int tn = 0; tn < 2; ++tn) {
            const int n = ncol0 + tn * 16;
#pragma unroll
            for (int e = 0; e < 8; ++e)
                out[(size_t)(mrow0 + mi * 16 + e) * N + n] = facc[mi][tn][e];
        }
}

// ---------------------------------------------------------------------------
// Fallback fused kernel (no workspace): dequant in-kernel.
// ---------------------------------------------------------------------------
__global__ __launch_bounds__(256)
void gemm_fused_kernel(const float* __restrict__ x,
                       const signed char* __restrict__ wq,
                       const float* __restrict__ scal,
                       float* __restrict__ out)
{
    __shared__ _Float16 Xs[BM * XSTR];
    __shared__ _Float16 Ws[BN * WSTR];

    const int tid    = threadIdx.x;
    const int lane   = tid & 31;
    const int wid    = tid >> 5;
    const int wave_m = wid >> 2;
    const int wave_n = wid & 3;
    const int l15    = lane & 15;
    const int lhalf  = (lane >> 4) & 1;
    const int koff   = lhalf * 8;

    const int bm = blockIdx.y * BM;
    const int bn = blockIdx.x * BN;

    v8f facc[2][2];
#pragma unroll
    for (int mi = 0; mi < 2; ++mi)
#pragma unroll
        for (int tn = 0; tn < 2; ++tn)
            facc[mi][tn] = (v8f){0.f, 0.f, 0.f, 0.f, 0.f, 0.f, 0.f, 0.f};

    for (int s = 0; s < NBLK; ++s) {
        const int kb = s * BK;
        float sc[2];
#pragma unroll
        for (int tn = 0; tn < 2; ++tn)
            sc[tn] = scal[s * N + bn + wave_n * 32 + tn * 16 + l15];

        __syncthreads();

#pragma unroll
        for (int i = 0; i < 8; ++i) {
            const int idx = tid + i * 256;
            const int row = idx >> 5;
            const int c4  = (idx & 31) * 4;
            const v4f v = *(const v4f*)(x + (size_t)(bm + row) * K + kb + c4);
            _Float16* dst = &Xs[row * XSTR + c4];
#pragma unroll
            for (int j = 0; j < 4; ++j) dst[j] = (_Float16)v[j];
        }
#pragma unroll
        for (int i = 0; i < 16; ++i) {
            const int idx = tid + i * 256;
            const int k   = idx >> 5;
            const int n4  = (idx & 31) * 4;
            const unsigned w32 =
                *(const unsigned*)(wq + (size_t)(kb + k) * N + bn + n4);
#pragma unroll
            for (int j = 0; j < 4; ++j) {
                const signed char b = (signed char)((w32 >> (8 * j)) & 0xff);
                Ws[(n4 + j) * WSTR + k] = (_Float16)b;
            }
        }
        __syncthreads();

        v8f bacc[2][2];
#pragma unroll
        for (int mi = 0; mi < 2; ++mi)
#pragma unroll
            for (int tn = 0; tn < 2; ++tn)
                bacc[mi][tn] = (v8f){0.f, 0.f, 0.f, 0.f, 0.f, 0.f, 0.f, 0.f};

#pragma unroll
        for (int ks = 0; ks < 4; ++ks) {
            const int kk = ks * 32;
            v16h A[2];
#pragma unroll
            for (int mi = 0; mi < 2; ++mi) {
                const int base = (wave_m * 32 + mi * 16 + l15) * XSTR + kk + koff;
                A[mi] = concat8(*(const v8h*)&Xs[base], *(const v8h*)&Xs[base + 16]);
            }
            v16h Bf[2];
#pragma unroll
            for (int tn = 0; tn < 2; ++tn) {
                const int base = (wave_n * 32 + tn * 16 + l15) * WSTR + kk + koff;
                Bf[tn] = concat8(*(const v8h*)&Ws[base], *(const v8h*)&Ws[base + 16]);
            }
#pragma unroll
            for (int mi = 0; mi < 2; ++mi)
#pragma unroll
                for (int tn = 0; tn < 2; ++tn)
                    bacc[mi][tn] = __builtin_amdgcn_wmma_f32_16x16x32_f16(
                        false, A[mi], false, Bf[tn],
                        (short)0, bacc[mi][tn], false, false);
        }

#pragma unroll
        for (int mi = 0; mi < 2; ++mi)
#pragma unroll
            for (int tn = 0; tn < 2; ++tn) {
                const float s_ = sc[tn];
#pragma unroll
                for (int e = 0; e < 8; ++e)
                    facc[mi][tn][e] = fmaf(s_, bacc[mi][tn][e], facc[mi][tn][e]);
            }
    }

    const int mrow0 = bm + wave_m * 32 + lhalf * 8;
    const int ncol0 = bn + wave_n * 32 + l15;
#pragma unroll
    for (int mi = 0; mi < 2; ++mi)
#pragma unroll
        for (int tn = 0; tn < 2; ++tn) {
            const int n = ncol0 + tn * 16;
#pragma unroll
            for (int e = 0; e < 8; ++e)
                out[(size_t)(mrow0 + mi * 16 + e) * N + n] = facc[mi][tn][e];
        }
}

extern "C" void kernel_launch(void* const* d_in, const int* in_sizes, int n_in,
                              void* d_out, int out_size, void* d_ws, size_t ws_size,
                              hipStream_t stream) {
    const float*       x  = (const float*)d_in[0];        // [2,1024,4096] f32
    const signed char* wq = (const signed char*)d_in[1];  // [32,128,4096] int8
    const float*       sc = (const float*)d_in[2];        // [32,4096] f32
    float*             o  = (float*)d_out;                // [2,1024,4096] f32

    const size_t need = (size_t)M * K * sizeof(_Float16)    // xh
                      + (size_t)N * K * sizeof(_Float16);   // whT

    dim3 grid(N / BN, M / BM);  // (32, 32)
    dim3 block(256);

    if (ws_size >= need) {
        _Float16* xh  = (_Float16*)d_ws;
        _Float16* whT = xh + (size_t)M * K;

        cvt_x_kernel<<<dim3((M * (size_t)K) / 4096), block, 0, stream>>>(x, xh);
        cvt_w_kernel<<<dim3(K / 128, N / 128), block, 0, stream>>>(wq, whT);
        gemm_f16_kernel<<<grid, block, 0, stream>>>(xh, whT, sc, o);
    } else {
        gemm_fused_kernel<<<grid, block, 0, stream>>>(x, wq, sc, o);
    }
}
